// FactorizedSpatioTemporalBlock_83099027243666
// MI455X (gfx1250) — matrix operand order repaired and checked by reading, not compiled, not run on hardware
//
#include <hip/hip_runtime.h>
#include <cstdint>
#include <cstddef>

#define DIM 512
#define HEADS 8
#define DHEAD 64
#define NTOK (2*16*24*24)   /* 18432 tokens, flat (b t) (h w) order */
#define FF_IN 1365
#define FF2X  2730
#define NCTX (32*77)
#define KVPAD 96
#define TSEQ 16
#define SPOS 576            /* 24*24 */

typedef __attribute__((ext_vector_type(16))) __bf16 v16bf;
typedef __attribute__((ext_vector_type(8)))  float  v8f;
typedef __attribute__((ext_vector_type(4))) unsigned int u32x4;
typedef __attribute__((ext_vector_type(4))) int i32x4;
typedef __attribute__((ext_vector_type(8))) int i32x8;

// ---------------- bf16 helpers (round-to-nearest-even) ----------------
__device__ __forceinline__ __bf16 f2bf(float f) {
  union { float f; uint32_t u; } a; a.f = f;
  uint32_t r = a.u + 0x7FFFu + ((a.u >> 16) & 1u);
  uint16_t h = (uint16_t)(r >> 16);
  return __builtin_bit_cast(__bf16, h);
}
__device__ __forceinline__ float bf2f(__bf16 b) {
  uint16_t h = __builtin_bit_cast(uint16_t, b);
  union { uint32_t u; float f; } a; a.u = ((uint32_t)h) << 16;
  return a.f;
}

// ---------------- Tensor Data Mover: 2D bf16 tile -> padded LDS ---------
// Packs D# group0/group1 per CDNA5 ISA:
//  g0: count=1 user D#, lds_addr, global tile addr, type=2 ("image")
//  g1: data_size=2B, pad_enable, pad_interval 2^(c+1) DWORDs, pad_amount c+1
//      DWORDs, tensor_dim0/1, tile_dim0/1, tensor_dim0_stride (elements).
// Issued once per wave (EXEC-ignored DMA); caller guards to a single wave and
// completes with s_wait_tensorcnt 0 before the publishing barrier.
__device__ __forceinline__ void tdm_load_2d_to_lds(
    const void* gptr, unsigned lds_off,
    unsigned tile_d0, unsigned tile_d1,
    unsigned long long row_stride_elems,
    unsigned pad_interval_code, unsigned pad_amount_code,
    unsigned tensor_d0, unsigned tensor_d1)
{
  unsigned long long ga = (unsigned long long)(size_t)gptr;
  u32x4 g0;
  g0[0] = 1u;                                                  // count=1
  g0[1] = lds_off;                                             // lds_addr
  g0[2] = (unsigned)ga;                                        // global[31:0]
  g0[3] = ((unsigned)(ga >> 32) & 0x01FFFFFFu) | (2u << 30);   // [56:32]|type=2
  unsigned w[8];
  w[0] = (1u << 16)                      // data_size = 2 bytes
       | (1u << 20)                      // pad_enable
       | (pad_interval_code << 22)
       | (pad_amount_code << 25);
  w[1] = (tensor_d0 & 0xFFFFu) << 16;                               // dim0[15:0]
  w[2] = ((tensor_d0 >> 16) & 0xFFFFu) | ((tensor_d1 & 0xFFFFu) << 16);
  w[3] = ((tensor_d1 >> 16) & 0xFFFFu) | ((tile_d0 & 0xFFFFu) << 16);
  w[4] = (tile_d1 & 0xFFFFu);                                       // tile_dim2=0
  w[5] = (unsigned)(row_stride_elems & 0xFFFFFFFFull);
  w[6] = (unsigned)((row_stride_elems >> 32) & 0xFFFFull);          // dim1_stride=0
  w[7] = 0u;
  i32x8 g1;
  for (int i = 0; i < 8; ++i) g1[i] = (int)w[i];
  i32x4 z4 = {0, 0, 0, 0};
#if defined(__clang_major__) && (__clang_major__ >= 23)
  i32x8 z8 = {0, 0, 0, 0, 0, 0, 0, 0};
  __builtin_amdgcn_tensor_load_to_lds(g0, g1, z4, z4, z8, 0);
#else
  __builtin_amdgcn_tensor_load_to_lds(g0, g1, z4, z4, 0);
#endif
}

// ---------------- WMMA fragment loaders (wave32 layouts, CDNA5 ISA) ----
// A 16x32 bf16: lanes 0-15 row=lane, K {0..7}+{16..23}; lanes 16-31 same row,
// K {8..15}+{24..31}.  Source is row-major [16][stride] bf16, 16B-aligned rows.
__device__ __forceinline__ v16bf ldA_frag(const __bf16* base, int stride, int lane) {
  int r  = lane & 15;
  int k0 = (lane >> 4) * 8;
  union { v16bf v; uint4 u[2]; } t;
  const __bf16* p = base + r * stride;
  t.u[0] = *(const uint4*)(p + k0);
  t.u[1] = *(const uint4*)(p + k0 + 16);
  return t.v;
}
// B 32x16 bf16: lane n holds column n; lanes 0-15 K {0..7}+{16..23}, lanes 16-31 K {8..15}+{24..31}.
// Source is B^T staged row-major: bT[n][k], stride multiple of 8 elems.
__device__ __forceinline__ v16bf ldB_frag(const __bf16* base, int stride, int lane) {
  int c  = lane & 15;
  int k0 = (lane >> 4) * 8;
  union { v16bf v; uint4 u[2]; } t;
  const __bf16* p = base + c * stride;
  t.u[0] = *(const uint4*)(p + k0);
  t.u[1] = *(const uint4*)(p + k0 + 16);
  return t.v;
}

// ---------------- GEMM: C[M,N] = A[M,K](bf16) x B[K,N](bf16) ------------
// AMODE 0: plain bf16 A (TDM-staged when tile fully interior).
// AMODE 1: GEGLU, A holds [u | gate]; staged tile = gelu(gate)*u.
// EPI 0: store f32.  EPI 1: store f32 + residual (may alias).  EPI 2: store bf16.
#define BM 64
#define BN 64
#define BKK 32
#define SAS 40
#define SBS 40

template<int AMODE, int EPI>
__global__ __launch_bounds__(128)
void gemm_bf16_wmma(const __bf16* __restrict__ A, int lda,
                    const __bf16* __restrict__ B, int ldb,
                    float* __restrict__ outf, const float* __restrict__ resf,
                    __bf16* __restrict__ outb, int ldo,
                    int M, int N, int K)
{
  __shared__ alignas(16) __bf16 sA[BM][SAS];
  __shared__ alignas(16) __bf16 sB[BN][SBS];   // transposed: sB[n][k]
  const int tid  = threadIdx.x;
  const int lane = tid & 31;
  const int wave = tid >> 5;
  const int m0 = blockIdx.y * BM;
  const int n0 = blockIdx.x * BN;
  const bool avec = ((lda & 7) == 0);
  const bool bvec = ((ldb & 7) == 0);

  v8f acc[4];
  for (int g = 0; g < 4; ++g)
    for (int i = 0; i < 8; ++i) acc[g][i] = 0.f;

  for (int k0 = 0; k0 < K; k0 += BKK) {
    __syncthreads();
    bool tdm_a = false;
    // ---- stage A tile (64x32, row stride SAS=40 = 16 DWORDs + 4 DWORD pad)
    if constexpr (AMODE == 0) {
      tdm_a = avec && (m0 + BM <= M) && (k0 + BKK <= K);
      if (tdm_a) {
        if (wave == 0)
          tdm_load_2d_to_lds(A + (size_t)m0 * lda + k0,
                             (unsigned)(size_t)&sA[0][0],
                             /*tile_d0=*/BKK, /*tile_d1=*/BM,
                             (unsigned long long)lda,
                             /*pad_interval 2^(3+1)=16 DW*/3u,
                             /*pad_amount 4 DW*/3u,
                             (unsigned)K, (unsigned)M);
      } else {
        for (int it = 0; it < 2; ++it) {
          int idx = it * 128 + tid;            // 0..255 -> 64 rows x 4 segs
          int r = idx >> 2, seg = (idx & 3) * 8;
          int gm = m0 + r, gk = k0 + seg;
          uint4 val{0u, 0u, 0u, 0u};
          if (gm < M) {
            if (avec && gk + 7 < K) {
              val = *(const uint4*)(A + (size_t)gm * lda + gk);
            } else {
              __bf16 tmp[8];
              for (int u = 0; u < 8; ++u)
                tmp[u] = (gk + u < K) ? A[(size_t)gm * lda + gk + u] : f2bf(0.f);
              __builtin_memcpy(&val, tmp, 16);
            }
          }
          *(uint4*)(&sA[r][seg]) = val;
        }
      }
    } else {
      for (int it = 0; it < 16; ++it) {
        int idx = it * 128 + tid;            // 0..2047
        int r = idx >> 5, c = idx & 31;
        int gm = m0 + r, gk = k0 + c;
        float v = 0.f;
        if (gm < M && gk < K) {
          float u    = bf2f(A[(size_t)gm * lda + gk]);
          float gate = bf2f(A[(size_t)gm * lda + FF_IN + gk]);
          v = 0.5f * gate * (1.f + erff(gate * 0.70710678118f)) * u; // exact GELU
        }
        sA[r][c] = f2bf(v);
      }
    }
    // ---- stage B tile transposed: sB[n][k] = B[k0+k][n0+n] ----
    for (int it = 0; it < 2; ++it) {
      int idx = it * 128 + tid;              // 0..255 -> 32 k x 8 n-segs
      int kk = idx >> 3, nseg = (idx & 7) * 8;
      int gk = k0 + kk;
      if (bvec && gk < K && n0 + nseg + 7 < N) {
        uint4 val = *(const uint4*)(B + (size_t)gk * ldb + n0 + nseg);
        const __bf16* p = (const __bf16*)&val;
        for (int u = 0; u < 8; ++u) sB[nseg + u][kk] = p[u];
      } else {
        for (int u = 0; u < 8; ++u) {
          int gn = n0 + nseg + u;
          sB[nseg + u][kk] = (gk < K && gn < N) ? B[(size_t)gk * ldb + gn] : f2bf(0.f);
        }
      }
    }
    // prefetch next A K-tile (gfx1250 global_prefetch_b8; speculative-safe)
    {
      int pr = m0 + (tid & 63);
      if (pr < M && k0 + BKK < K)
        __builtin_prefetch(A + (size_t)pr * lda + k0 + BKK, 0, 0);
    }
    if constexpr (AMODE == 0) {
      if (tdm_a && wave == 0) __builtin_amdgcn_s_wait_tensorcnt(0);
    }
    __syncthreads();
    // ---- compute: each wave does one 16-row strip x four 16-col groups ----
    v16bf af = ldA_frag(&sA[wave * 16][0], SAS, lane);
    for (int g = 0; g < 4; ++g) {
      v16bf bf = ldB_frag(&sB[g * 16][0], SBS, lane);
      acc[g] = __builtin_amdgcn_wmma_f32_16x16x32_bf16(
          false, af, false, bf, (short)0, acc[g], false, false);
    }
  }
  // ---- epilogue: C/D layout element (vj,lane) -> m=vj+8*(lane>=16), n=lane&15
  int half = lane >> 4, nn = lane & 15;
  for (int g = 0; g < 4; ++g)
    for (int vj = 0; vj < 8; ++vj) {
      int gm = m0 + wave * 16 + vj + 8 * half;
      int gn = n0 + g * 16 + nn;
      if (gm < M && gn < N) {
        float v = acc[g][vj];
        if constexpr (EPI == 0)      outf[(size_t)gm * ldo + gn] = v;
        else if constexpr (EPI == 1) outf[(size_t)gm * ldo + gn] = v + resf[(size_t)gm * ldo + gn];
        else                         outb[(size_t)gm * ldo + gn] = f2bf(v);
      }
    }
}

// ---------------- Flash attention (WMMA QK^T and PV) --------------------
// MODE 0: spatial self-attn  (batch=bt=32, n=576, token-layout K/V)
// MODE 1: cross-attn         (K/V = [b][h][KVPAD][64] assembled buffer)
// MODE 2: temporal causal+ALiBi (batch=(b,pos)=1152, seq stride 576*512)
template<int MODE>
__global__ __launch_bounds__(128)
void attn_wmma(const __bf16* __restrict__ Q, const __bf16* __restrict__ Kp,
               const __bf16* __restrict__ Vp, __bf16* __restrict__ Out,
               int n, int kvn, int kvn_pad)
{
  __shared__ alignas(16) __bf16 sK[32][72];      // [j][d]   (= B^T for Q.K^T)
  __shared__ alignas(16) __bf16 sV[64][40];      // [d][j]   (= B^T for P.V)
  __shared__ alignas(16) __bf16 sP[4][16][40];   // per-wave P re-fragment
  const int tid = threadIdx.x, lane = tid & 31, wave = tid >> 5;
  const int h = blockIdx.y;
  const int bz = blockIdx.z;
  const int q0 = blockIdx.x * 64 + wave * 16;
  const bool active = (q0 < n);

  size_t qbase, qrstride;
  if constexpr (MODE == 2) {
    int b = bz / SPOS, pos = bz % SPOS;
    qbase = ((size_t)b * TSEQ * SPOS + pos) * DIM + (size_t)h * DHEAD;
    qrstride = (size_t)SPOS * DIM;              // step along t
  } else {
    qbase = (size_t)bz * n * DIM + (size_t)h * DHEAD;
    qrstride = DIM;
  }
  size_t kbase, krstride;
  if constexpr (MODE == 1) {
    kbase = ((size_t)(bz * HEADS + h)) * kvn_pad * DHEAD;
    krstride = DHEAD;
  } else { kbase = qbase; krstride = qrstride; }

  // Q fragments for the two 32-wide d-chunks, loaded straight from global.
  v16bf qf[2];
  if (active) {
    for (int dc = 0; dc < 2; ++dc) {
      int r = lane & 15, k0 = (lane >> 4) * 8;
      const __bf16* qp = Q + qbase + (size_t)(q0 + r) * qrstride + dc * 32;
      union { v16bf v; uint4 u[2]; } t;
      t.u[0] = *(const uint4*)(qp + k0);
      t.u[1] = *(const uint4*)(qp + k0 + 16);
      qf[dc] = t.v;
    }
  }
  float m_[8], l_[8];
  for (int i = 0; i < 8; ++i) { m_[i] = -3.4e38f; l_[i] = 0.f; }
  v8f o[4];
  for (int g = 0; g < 4; ++g)
    for (int i = 0; i < 8; ++i) o[g][i] = 0.f;
  const float slope = exp2f(-(float)(h + 1));   // ALiBi, heads=8 -> 0.5^(h+1)

  for (int kv0 = 0; kv0 < kvn_pad; kv0 += 32) {
    __syncthreads();
    // ---- stage K chunk row-major [32][64] (72-elem stride = 32+4 DWORDs) --
    const bool full = (kv0 + 32 <= kvn);
    if (full) {
      if (wave == 0)
        tdm_load_2d_to_lds(Kp + kbase + (size_t)kv0 * krstride,
                           (unsigned)(size_t)&sK[0][0],
                           /*tile_d0=*/64u, /*tile_d1=*/32u,
                           (unsigned long long)krstride,
                           /*pad_interval 2^(4+1)=32 DW*/4u,
                           /*pad_amount 4 DW*/3u,
                           64u, 32u);
    } else {
      for (int it = 0; it < 2; ++it) {
        int idx = it * 128 + tid;               // 0..255
        int j = idx >> 3, seg = (idx & 7) * 8;
        int gj = kv0 + j;
        uint4 kv{0u,0u,0u,0u};
        if (gj < kvn) kv = *(const uint4*)(Kp + kbase + (size_t)gj * krstride + seg);
        *(uint4*)(&sK[j][seg]) = kv;
      }
    }
    // ---- stage V chunk transposed [64][32] (manual: TDM cannot transpose) -
    for (int it = 0; it < 2; ++it) {
      int idx = it * 128 + tid;                 // 0..255
      int j = idx >> 3, seg = (idx & 7) * 8;
      int gj = kv0 + j;
      uint4 vv{0u,0u,0u,0u};
      if (gj < kvn) vv = *(const uint4*)(Vp + kbase + (size_t)gj * krstride + seg);
      const __bf16* p = (const __bf16*)&vv;
      for (int u = 0; u < 8; ++u) sV[seg + u][j] = p[u];
    }
    if (full && wave == 0) __builtin_amdgcn_s_wait_tensorcnt(0);
    __syncthreads();
    if (active) {
      // S(16x32) = Q(16x64) . K^T, two 16-col groups, K-dim in 2 chunks
      v8f S[2];
      for (int g = 0; g < 2; ++g) {
        for (int i = 0; i < 8; ++i) S[g][i] = 0.f;
        for (int dc = 0; dc < 2; ++dc) {
          int c = lane & 15, k0 = dc * 32 + (lane >> 4) * 8;
          union { v16bf v; uint4 u[2]; } t;
          const __bf16* kp = &sK[g * 16 + c][0];
          t.u[0] = *(const uint4*)(kp + k0);
          t.u[1] = *(const uint4*)(kp + k0 + 16);
          S[g] = __builtin_amdgcn_wmma_f32_16x16x32_bf16(
              false, qf[dc], false, t.v, (short)0, S[g], false, false);
        }
      }
      // online softmax: row = vj + 8*(lane>=16); cols across the 16-lane half
      int half = lane >> 4, nn = lane & 15;
      float pv0[8], pv1[8], alpha[8];
      for (int vj = 0; vj < 8; ++vj) {
        int ig = q0 + vj + 8 * half;
        float s0 = S[0][vj] * 8.f, s1 = S[1][vj] * 8.f;   // SCALE = 8
        int j0 = kv0 + nn, j1 = kv0 + 16 + nn;
        if constexpr (MODE == 2) {
          s0 += slope * -fabsf((float)(ig - j0));
          s1 += slope * -fabsf((float)(ig - j1));
          if (j0 > ig || j0 >= kvn) s0 = -3.4e38f;
          if (j1 > ig || j1 >= kvn) s1 = -3.4e38f;
        } else if constexpr (MODE == 1) {
          if (j0 >= kvn) s0 = -3.4e38f;
          if (j1 >= kvn) s1 = -3.4e38f;
        }
        float mx = fmaxf(s0, s1);
        for (int d = 1; d < 16; d <<= 1) mx = fmaxf(mx, __shfl_xor(mx, d, 16));
        float mnew = fmaxf(m_[vj], mx);
        float a  = __expf(m_[vj] - mnew);
        float p0 = __expf(s0 - mnew), p1 = __expf(s1 - mnew);
        float rs = p0 + p1;
        for (int d = 1; d < 16; d <<= 1) rs += __shfl_xor(rs, d, 16);
        l_[vj] = l_[vj] * a + rs;
        m_[vj] = mnew;
        alpha[vj] = a;
        pv0[vj] = p0; pv1[vj] = p1;
      }
      for (int g = 0; g < 4; ++g)
        for (int vj = 0; vj < 8; ++vj) o[g][vj] *= alpha[vj];
      // C-layout -> A-layout for P via per-wave LDS round trip
      for (int vj = 0; vj < 8; ++vj) {
        sP[wave][vj + 8 * half][nn]      = f2bf(pv0[vj]);
        sP[wave][vj + 8 * half][16 + nn] = f2bf(pv1[vj]);
      }
    }
    __syncthreads();
    if (active) {
      v16bf pf = ldA_frag(&sP[wave][0][0], 40, lane);
      for (int g = 0; g < 4; ++g) {
        v16bf vf = ldB_frag(&sV[g * 16][0], 40, lane);
        o[g] = __builtin_amdgcn_wmma_f32_16x16x32_bf16(
            false, pf, false, vf, (short)0, o[g], false, false);
      }
    }
  }
  if (active) {
    int half = lane >> 4, nn = lane & 15;
    for (int g = 0; g < 4; ++g)
      for (int vj = 0; vj < 8; ++vj) {
        int r = q0 + vj + 8 * half;
        if (r < n) {
          float denom = l_[vj];
          float v = (denom > 0.f) ? o[g][vj] / denom : 0.f;
          Out[qbase + (size_t)r * qrstride + g * 16 + nn] = f2bf(v);
        }
      }
  }
}

// ---------------- LayerNorm (C=512), one wave per row -------------------
template<bool OBF>
__global__ __launch_bounds__(128)
void layernorm_k(const float* __restrict__ X, const float* __restrict__ gma,
                 const float* __restrict__ beta, void* __restrict__ outp, int nrows)
{
  int wave = threadIdx.x >> 5, lane = threadIdx.x & 31;
  int row = blockIdx.x * 4 + wave;
  if (row >= nrows) return;
  const float* x = X + (size_t)row * DIM;
  float4 v[4];
  float s = 0.f, ss = 0.f;
  for (int i = 0; i < 4; ++i) {
    v[i] = *(const float4*)(x + i * 128 + lane * 4);
    s  += v[i].x + v[i].y + v[i].z + v[i].w;
    ss += v[i].x * v[i].x + v[i].y * v[i].y + v[i].z * v[i].z + v[i].w * v[i].w;
  }
  for (int d = 1; d < 32; d <<= 1) { s += __shfl_xor(s, d, 32); ss += __shfl_xor(ss, d, 32); }
  float mu = s * (1.f / 512.f);
  float var = ss * (1.f / 512.f) - mu * mu;
  float rstd = rsqrtf(fmaxf(var, 0.f) + 1e-5f);
  for (int i = 0; i < 4; ++i) {
    float e[4] = { v[i].x, v[i].y, v[i].z, v[i].w };
    for (int j = 0; j < 4; ++j) {
      int c = i * 128 + lane * 4 + j;
      float y = (e[j] - mu) * rstd * gma[c];
      if (beta) y += beta[c];
      if constexpr (OBF) ((__bf16*)outp)[(size_t)row * DIM + c] = f2bf(y);
      else               ((float*)outp)[(size_t)row * DIM + c]  = y;
    }
  }
}

// ---------------- per-head l2norm * scale -> bf16 -----------------------
__global__ __launch_bounds__(128)
void l2norm_head_k(const float* __restrict__ X, int in_ld,
                   const float* __restrict__ scale, __bf16* __restrict__ O, int nrows)
{
  int wid = blockIdx.x * 4 + (threadIdx.x >> 5);
  int lane = threadIdx.x & 31;
  if (wid >= nrows * HEADS) return;
  int row = wid >> 3, h = wid & 7;
  const float* x = X + (size_t)row * in_ld + h * DHEAD;
  float2 e = *(const float2*)(x + lane * 2);
  float ss = e.x * e.x + e.y * e.y;
  for (int d = 1; d < 32; d <<= 1) ss += __shfl_xor(ss, d, 32);
  float inv = 1.f / fmaxf(sqrtf(ss), 1e-12f);
  __bf16* o = O + (size_t)row * DIM + h * DHEAD;
  o[lane * 2]     = f2bf(e.x * inv * scale[lane * 2]);
  o[lane * 2 + 1] = f2bf(e.y * inv * scale[lane * 2 + 1]);
}

// ---------------- f32 -> bf16 conversions -------------------------------
__global__ void cvt_bf16_k(const float* __restrict__ X, __bf16* __restrict__ O, size_t n) {
  size_t i = (size_t)blockIdx.x * blockDim.x + threadIdx.x;
  if (i < n) O[i] = f2bf(X[i]);
}
__global__ void cvt_v_k(const float* __restrict__ X, __bf16* __restrict__ O, int nrows) {
  // X: [nrows][1024] (kv proj), take V = cols [512,1024) -> O [nrows][512]
  size_t i = (size_t)blockIdx.x * blockDim.x + threadIdx.x;
  size_t total = (size_t)nrows * DIM;
  if (i >= total) return;
  size_t r = i / DIM; int c = (int)(i % DIM);
  O[i] = f2bf(X[r * 1024 + 512 + c]);
}

// ---------------- cross-attn K/V assembly (null kv + ctx, l2norm K) -----
__global__ __launch_bounds__(128)
void cross_kv_k(const float* __restrict__ ctx_kv,  // [NCTX][1024]
                const float* __restrict__ nullkv,  // [8][4][64]
                const float* __restrict__ kscale,  // [64]
                __bf16* __restrict__ CK, __bf16* __restrict__ CV)
{
  int wid = blockIdx.x * 4 + (threadIdx.x >> 5);   // over 32*8*KVPAD rows
  int lane = threadIdx.x & 31;
  if (wid >= 32 * HEADS * KVPAD) return;
  int j = wid % KVPAD, h = (wid / KVPAD) % HEADS, b = wid / (KVPAD * HEADS);
  float k0 = 0.f, k1 = 0.f, v0 = 0.f, v1 = 0.f;
  if (j < 2) {
    const float* nk = nullkv + (size_t)(h * 4 + j * 2) * DHEAD;
    const float* nv = nk + DHEAD;
    k0 = nk[lane * 2]; k1 = nk[lane * 2 + 1];
    v0 = nv[lane * 2]; v1 = nv[lane * 2 + 1];
  } else if (j < 79) {
    const float* r = ctx_kv + (size_t)(b * 77 + (j - 2)) * 1024;
    k0 = r[h * DHEAD + lane * 2];       k1 = r[h * DHEAD + lane * 2 + 1];
    v0 = r[512 + h * DHEAD + lane * 2]; v1 = r[512 + h * DHEAD + lane * 2 + 1];
  }
  float ss = k0 * k0 + k1 * k1;
  for (int d = 1; d < 32; d <<= 1) ss += __shfl_xor(ss, d, 32);
  float inv = (j < 79) ? 1.f / fmaxf(sqrtf(ss), 1e-12f) : 0.f;
  size_t base = ((size_t)(b * HEADS + h) * KVPAD + j) * DHEAD;
  CK[base + lane * 2]     = f2bf(k0 * inv * kscale[lane * 2]);
  CK[base + lane * 2 + 1] = f2bf(k1 * inv * kscale[lane * 2 + 1]);
  CV[base + lane * 2]     = f2bf(v0);
  CV[base + lane * 2 + 1] = f2bf(v1);
}

// ---------------- host orchestration ------------------------------------
extern "C" void kernel_launch(void* const* d_in, const int* in_sizes, int n_in,
                              void* d_out, int out_size, void* d_ws, size_t ws_size,
                              hipStream_t stream)
{
  (void)in_sizes; (void)n_in; (void)out_size; (void)ws_size;
  const float* x      = (const float*)d_in[0];
  const float* ctx    = (const float*)d_in[1];
  const float* sp_g   = (const float*)d_in[2];
  const float* sp_wq  = (const float*)d_in[3];
  const float* sp_wkv = (const float*)d_in[4];
  const float* sp_qs  = (const float*)d_in[5];
  const float* sp_ks  = (const float*)d_in[6];
  const float* sp_wo  = (const float*)d_in[7];
  const float* ca_g   = (const float*)d_in[8];
  const float* ca_cg  = (const float*)d_in[9];
  const float* ca_wq  = (const float*)d_in[10];
  const float* ca_wkv = (const float*)d_in[11];
  const float* ca_qs  = (const float*)d_in[12];
  const float* ca_ks  = (const float*)d_in[13];
  const float* ca_wo  = (const float*)d_in[14];
  const float* ca_nul = (const float*)d_in[15];
  const float* tp_g   = (const float*)d_in[16];
  const float* tp_wq  = (const float*)d_in[17];
  const float* tp_wkv = (const float*)d_in[18];
  const float* tp_qs  = (const float*)d_in[19];
  const float* tp_ks  = (const float*)d_in[20];
  const float* tp_wo  = (const float*)d_in[21];
  const float* ff_g   = (const float*)d_in[22];
  const float* ff_b   = (const float*)d_in[23];
  const float* ff_w1  = (const float*)d_in[24];
  const float* ff_w2  = (const float*)d_in[25];
  const float* out_g  = (const float*)d_in[26];

  char* ws = (char*)d_ws;
  size_t off = 0;
  auto alloc = [&](size_t b) { size_t o = off; off += (b + 255) & ~(size_t)255; return o; };
  float*  RES  = (float*) (ws + alloc((size_t)NTOK * DIM * 4));
  __bf16* ABUF = (__bf16*)(ws + alloc((size_t)NTOK * DIM * 2));
  __bf16* QB   = (__bf16*)(ws + alloc((size_t)NTOK * DIM * 2));
  __bf16* KB   = (__bf16*)(ws + alloc((size_t)NTOK * DIM * 2));
  __bf16* VB   = (__bf16*)(ws + alloc((size_t)NTOK * DIM * 2));
  float*  QF   = (float*) (ws + alloc((size_t)NTOK * DIM * 4));
  float*  KVF  = (float*) (ws + alloc((size_t)NTOK * 1024 * 4));  // adjacent to QF
  __bf16* FF1  = (__bf16*)QF;   // aliases QF+KVF (113 MB) once projections are dead
  __bf16* CTXA = (__bf16*)(ws + alloc((size_t)NCTX * DIM * 2));
  float*  CTXKV= (float*) (ws + alloc((size_t)NCTX * 1024 * 4));
  __bf16* CK   = (__bf16*)(ws + alloc((size_t)32 * HEADS * KVPAD * DHEAD * 2));
  __bf16* CV   = (__bf16*)(ws + alloc((size_t)32 * HEADS * KVPAD * DHEAD * 2));
  __bf16* Wq_s = (__bf16*)(ws + alloc((size_t)512 * 512 * 2));
  __bf16* Wkv_s= (__bf16*)(ws + alloc((size_t)512 * 1024 * 2));
  __bf16* Wo_s = (__bf16*)(ws + alloc((size_t)512 * 512 * 2));
  __bf16* Wq_c = (__bf16*)(ws + alloc((size_t)512 * 512 * 2));
  __bf16* Wkv_c= (__bf16*)(ws + alloc((size_t)512 * 1024 * 2));
  __bf16* Wo_c = (__bf16*)(ws + alloc((size_t)512 * 512 * 2));
  __bf16* Wq_t = (__bf16*)(ws + alloc((size_t)512 * 512 * 2));
  __bf16* Wkv_t= (__bf16*)(ws + alloc((size_t)512 * 1024 * 2));
  __bf16* Wo_t = (__bf16*)(ws + alloc((size_t)512 * 512 * 2));
  __bf16* W1   = (__bf16*)(ws + alloc((size_t)512 * FF2X * 2));
  __bf16* W2   = (__bf16*)(ws + alloc((size_t)FF_IN * 512 * 2));

  auto cvt = [&](const float* s, __bf16* dst, size_t n) {
    cvt_bf16_k<<<dim3((unsigned)((n + 255) / 256)), dim3(256), 0, stream>>>(s, dst, n);
  };
  cvt(sp_wq, Wq_s, 512 * 512);   cvt(sp_wkv, Wkv_s, 512 * 1024); cvt(sp_wo, Wo_s, 512 * 512);
  cvt(ca_wq, Wq_c, 512 * 512);   cvt(ca_wkv, Wkv_c, 512 * 1024); cvt(ca_wo, Wo_c, 512 * 512);
  cvt(tp_wq, Wq_t, 512 * 512);   cvt(tp_wkv, Wkv_t, 512 * 1024); cvt(tp_wo, Wo_t, 512 * 512);
  cvt(ff_w1, W1, (size_t)512 * FF2X); cvt(ff_w2, W2, (size_t)FF_IN * 512);

  const dim3 blk(128);
  const dim3 lnG((NTOK + 3) / 4);
  const dim3 l2G((NTOK * HEADS + 3) / 4);
  auto gemmGrid = [](int M, int N) { return dim3((N + 63) / 64, (M + 63) / 64); };

  // ===== Stage A: spatial self-attention =====
  layernorm_k<true><<<lnG, blk, 0, stream>>>(x, sp_g, nullptr, ABUF, NTOK);
  gemm_bf16_wmma<0,0><<<gemmGrid(NTOK, 512), blk, 0, stream>>>(
      ABUF, 512, Wq_s, 512, QF, nullptr, nullptr, 512, NTOK, 512, 512);
  gemm_bf16_wmma<0,0><<<gemmGrid(NTOK, 1024), blk, 0, stream>>>(
      ABUF, 512, Wkv_s, 1024, KVF, nullptr, nullptr, 1024, NTOK, 1024, 512);
  l2norm_head_k<<<l2G, blk, 0, stream>>>(QF, 512, sp_qs, QB, NTOK);
  l2norm_head_k<<<l2G, blk, 0, stream>>>(KVF, 1024, sp_ks, KB, NTOK);
  cvt_v_k<<<dim3(((size_t)NTOK * DIM + 255) / 256), dim3(256), 0, stream>>>(KVF, VB, NTOK);
  attn_wmma<0><<<dim3(9, HEADS, 32), blk, 0, stream>>>(QB, KB, VB, ABUF, SPOS, SPOS, SPOS);
  gemm_bf16_wmma<0,1><<<gemmGrid(NTOK, 512), blk, 0, stream>>>(
      ABUF, 512, Wo_s, 512, RES, x, nullptr, 512, NTOK, 512, 512);

  // ===== Stage B: spatial cross-attention =====
  layernorm_k<true><<<lnG, blk, 0, stream>>>(RES, ca_g, nullptr, ABUF, NTOK);
  layernorm_k<true><<<dim3((NCTX + 3) / 4), blk, 0, stream>>>(ctx, ca_cg, nullptr, CTXA, NCTX);
  gemm_bf16_wmma<0,0><<<gemmGrid(NTOK, 512), blk, 0, stream>>>(
      ABUF, 512, Wq_c, 512, QF, nullptr, nullptr, 512, NTOK, 512, 512);
  gemm_bf16_wmma<0,0><<<gemmGrid(NCTX, 1024), blk, 0, stream>>>(
      CTXA, 512, Wkv_c, 1024, CTXKV, nullptr, nullptr, 1024, NCTX, 1024, 512);
  l2norm_head_k<<<l2G, blk, 0, stream>>>(QF, 512, ca_qs, QB, NTOK);
  cross_kv_k<<<dim3((32 * HEADS * KVPAD + 3) / 4), blk, 0, stream>>>(CTXKV, ca_nul, ca_ks, CK, CV);
  attn_wmma<1><<<dim3(9, HEADS, 32), blk, 0, stream>>>(QB, CK, CV, ABUF, SPOS, 79, KVPAD);
  gemm_bf16_wmma<0,1><<<gemmGrid(NTOK, 512), blk, 0, stream>>>(
      ABUF, 512, Wo_c, 512, RES, RES, nullptr, 512, NTOK, 512, 512);

  // ===== Stage C: temporal causal self-attention (ALiBi) =====
  layernorm_k<true><<<lnG, blk, 0, stream>>>(RES, tp_g, nullptr, ABUF, NTOK);
  gemm_bf16_wmma<0,0><<<gemmGrid(NTOK, 512), blk, 0, stream>>>(
      ABUF, 512, Wq_t, 512, QF, nullptr, nullptr, 512, NTOK, 512, 512);
  gemm_bf16_wmma<0,0><<<gemmGrid(NTOK, 1024), blk, 0, stream>>>(
      ABUF, 512, Wkv_t, 1024, KVF, nullptr, nullptr, 1024, NTOK, 1024, 512);
  l2norm_head_k<<<l2G, blk, 0, stream>>>(QF, 512, tp_qs, QB, NTOK);
  l2norm_head_k<<<l2G, blk, 0, stream>>>(KVF, 1024, tp_ks, KB, NTOK);
  cvt_v_k<<<dim3(((size_t)NTOK * DIM + 255) / 256), dim3(256), 0, stream>>>(KVF, VB, NTOK);
  attn_wmma<2><<<dim3(1, HEADS, 2 * SPOS), blk, 0, stream>>>(QB, KB, VB, ABUF, TSEQ, TSEQ, 32);
  gemm_bf16_wmma<0,1><<<gemmGrid(NTOK, 512), blk, 0, stream>>>(
      ABUF, 512, Wo_t, 512, RES, RES, nullptr, 512, NTOK, 512, 512);

  // ===== Stage D: GEGLU feed-forward =====
  layernorm_k<true><<<lnG, blk, 0, stream>>>(RES, ff_g, ff_b, ABUF, NTOK);
  gemm_bf16_wmma<0,2><<<gemmGrid(NTOK, FF2X), blk, 0, stream>>>(
      ABUF, 512, W1, FF2X, nullptr, nullptr, FF1, FF2X, NTOK, FF2X, 512);
  gemm_bf16_wmma<1,1><<<gemmGrid(NTOK, 512), blk, 0, stream>>>(
      FF1, FF2X, W2, 512, RES, RES, nullptr, 512, NTOK, 512, FF_IN);

  // ===== final LayerNorm -> f32 output =====
  layernorm_k<false><<<lnG, blk, 0, stream>>>(RES, out_g, nullptr, d_out, NTOK);
}
